// RedecodeDecoder_68788196213000
// MI455X (gfx1250) — matrix operand order repaired and checked by reading, not compiled, not run on hardware
//
#include <hip/hip_runtime.h>
#include <hip/hip_bf16.h>

// ---------------- constants from the reference ----------------
#define VSZ   16000
#define EDIM  512
#define HDIM  1024
#define BSZ   32
#define RLEN  50
#define TSTEPS 50
#define T1    51          // TSTEPS + 1
#define G3    (3*HDIM)    // 3072
#define START_TOK 1
#define END_TOK   2

typedef __bf16 bf16_t;
typedef __attribute__((ext_vector_type(16))) __bf16 v16bf;
typedef __attribute__((ext_vector_type(8)))  __bf16 v8bf;
typedef __attribute__((ext_vector_type(8)))  float  v8f;

// ============================================================
// WMMA GEMM:  C[M,N] = A[M,K] * W[N,K]^T + bias[N]
// A bf16 row-major (lda), W bf16 row-major [N,K] (ldw=K),
// C fp32 with row stride ldc. M is always 32 (grid.y = 2).
// Fragment layouts per CDNA5 ISA 7.12.2 (wave32).
// ============================================================
__device__ __forceinline__ v16bf load_a_frag(const bf16_t* __restrict__ A,
                                             int lda, int m, int kc, int hi) {
  const bf16_t* p = A + (size_t)m * lda + kc;
  v8bf lo = *(const v8bf*)(p + hi * 8);        // K = hi*8 .. hi*8+7
  v8bf hv = *(const v8bf*)(p + 16 + hi * 8);   // K = 16+hi*8 .. 16+hi*8+7
  v16bf r;
#pragma unroll
  for (int i = 0; i < 8; ++i) { r[i] = lo[i]; r[i + 8] = hv[i]; }
  return r;
}

__device__ __forceinline__ v16bf load_b_frag(const bf16_t* __restrict__ W,
                                             int ldw, int n, int kc, int hi) {
  const bf16_t* p = W + (size_t)n * ldw + kc + hi * 8; // K = hi*8 .. hi*8+7, 16+hi*8 .. 16+hi*8+7
  v8bf b0 = *(const v8bf*)(p);
  v8bf b1 = *(const v8bf*)(p + 16);
  v16bf r;
#pragma unroll
  for (int i = 0; i < 8; ++i) { r[i] = b0[i]; r[i + 8] = b1[i]; }
  return r;
}

struct GemmArgs {
  const bf16_t* A; int lda;
  const bf16_t* W; int ldw;
  const float*  bias;
  float*        C; long ldc;
  int K;
};

__device__ __forceinline__ void gemm_body(const GemmArgs g) {
  const int lane = threadIdx.x & 31;
  const int wave = threadIdx.x >> 5;
  const int n0 = (blockIdx.x * 4 + wave) * 16;
  const int m0 = blockIdx.y * 16;
  const int hi = lane >> 4;
  const int lm = lane & 15;

  v8f acc = {};
  for (int kc = 0; kc < g.K; kc += 32) {
    v16bf a = load_a_frag(g.A, g.lda, m0 + lm, kc, hi);
    v16bf b = load_b_frag(g.W, g.ldw, n0 + lm, kc, hi);
    acc = __builtin_amdgcn_wmma_f32_16x16x32_bf16(
        false, a, false, b, (short)0, acc, false, false);
  }

  const int n = n0 + lm;
  const float bv = g.bias ? g.bias[n] : 0.0f;
#pragma unroll
  for (int r = 0; r < 8; ++r) {
    const int m = m0 + r + 8 * hi;   // D layout: VGPR r -> M=r (lanes<16) / M=8+r
    g.C[(size_t)m * g.ldc + n] = acc[r] + bv;
  }
}

__global__ void __launch_bounds__(128)
wmma_gemm_bias(GemmArgs g) { gemm_body(g); }

// two independent GEMMs with identical N, selected by blockIdx.z
__global__ void __launch_bounds__(128)
wmma_gemm_bias2(GemmArgs g0, GemmArgs g1) { gemm_body(blockIdx.z == 0 ? g0 : g1); }

// ============================================================
// Small helper kernels
// ============================================================
__global__ void cvt_f32_bf16(const float* __restrict__ s, bf16_t* __restrict__ d, int n) {
  int i = blockIdx.x * 256 + threadIdx.x;
  if (i < n) d[i] = (bf16_t)s[i];
}

__global__ void copy_f32(const float* __restrict__ s, float* __restrict__ d, int n) {
  int i = blockIdx.x * 256 + threadIdx.x;
  if (i < n) d[i] = s[i];
}

__global__ void zero_f32(float* __restrict__ d, int n) {
  int i = blockIdx.x * 256 + threadIdx.x;
  if (i < n) d[i] = 0.0f;
}

__global__ void set_int(int* __restrict__ d, int v, int n) {
  int i = blockIdx.x * 256 + threadIdx.x;
  if (i < n) d[i] = v;
}

__global__ void set_last_end(int* __restrict__ ptok) {
  int b = threadIdx.x;
  if (b < BSZ) ptok[b * T1 + TSTEPS] = END_TOK;
}

// x[b, 0..E) = embedding_bf16[tok[b*tstride]]
__global__ void embed_bf16(const bf16_t* __restrict__ emb,
                           const int* __restrict__ tok, int tstride,
                           bf16_t* __restrict__ x, int ldx) {
  int i = blockIdx.x * 256 + threadIdx.x;
  int b = i / EDIM, e = i % EDIM;
  if (b < BSZ) x[(size_t)b * ldx + e] = emb[(size_t)tok[b * tstride] * EDIM + e];
}

// h bf16 copy (used only for phase-start priming)
__global__ void h_to_bf16(const float* __restrict__ h, bf16_t* __restrict__ hb) {
  int i = blockIdx.x * 256 + threadIdx.x;
  hb[i] = (bf16_t)h[i];
}

// ============================================================
// Fused attention: scores + masked softmax + context (bf16 -> x slice)
// one 256-thread block (8 waves) per batch row; wave32 shuffles.
// ============================================================
__global__ void __launch_bounds__(256)
fused_attend(const float* __restrict__ q, const float* __restrict__ v,
             const int* __restrict__ imask, const int* __restrict__ plen,
             int R, bf16_t* __restrict__ x, int ldx, int off) {
  const int b = blockIdx.x;
  const int lane = threadIdx.x & 31;
  const int wave = threadIdx.x >> 5;
  __shared__ float sw[64];

  const float* qp = q + (size_t)b * HDIM;
  // ---- scores: each wave handles r = wave, wave+8, ...
  for (int r = wave; r < R; r += 8) {
    const float* vp = v + ((size_t)b * R + r) * HDIM;
    float acc = 0.0f;
    for (int i = lane; i < HDIM; i += 32) acc += qp[i] * vp[i];
#pragma unroll
    for (int o = 16; o > 0; o >>= 1) acc += __shfl_down(acc, o, 32);
    if (lane == 0) {
      bool ok = imask ? (imask[b * R + r] > 0) : (r < plen[b]);
      sw[r] = ok ? acc : -1.0e9f;
    }
  }
  if ((int)threadIdx.x >= R && threadIdx.x < 64) sw[threadIdx.x] = -3.0e38f;
  __syncthreads();

  // ---- softmax over R (<=64) done by wave 0, two entries per lane
  if (wave == 0) {
    float v0 = sw[lane], v1 = sw[lane + 32];
    float m = fmaxf(v0, v1);
#pragma unroll
    for (int o = 16; o > 0; o >>= 1) m = fmaxf(m, __shfl_down(m, o, 32));
    m = __shfl(m, 0, 32);
    float e0 = (lane < R)      ? __expf(v0 - m) : 0.0f;
    float e1 = (lane + 32 < R) ? __expf(v1 - m) : 0.0f;
    float s = e0 + e1;
#pragma unroll
    for (int o = 16; o > 0; o >>= 1) s += __shfl_down(s, o, 32);
    s = __shfl(s, 0, 32);
    if (lane < R)      sw[lane]      = e0 / s;
    if (lane + 32 < R) sw[lane + 32] = e1 / s;
  }
  __syncthreads();

  // ---- context: ctx[b,h] = sum_r w[r] * v[b,r,h], written bf16 into x slice
  for (int h = threadIdx.x; h < HDIM; h += 256) {
    float acc = 0.0f;
    for (int r = 0; r < R; ++r) acc += sw[r] * v[((size_t)b * R + r) * HDIM + h];
    x[(size_t)b * ldx + off + h] = (bf16_t)acc;
  }
}

// ============================================================
// Fused GRU gates: writes new h (f32) AND its bf16 copy.
// If plen != null, keep old h where !(t < plen[b]).
// ============================================================
__global__ void gru_gates(const float* __restrict__ gi, const float* __restrict__ gh,
                          const float* __restrict__ h, float* __restrict__ hout,
                          bf16_t* __restrict__ hbf,
                          const int* __restrict__ plen, int t) {
  int i = blockIdx.x * 256 + threadIdx.x;
  int b = i / HDIM, j = i % HDIM;
  if (b >= BSZ) return;
  size_t g = (size_t)b * G3 + j;
  float ir = gi[g], iz = gi[g + HDIM], in = gi[g + 2 * HDIM];
  float hr = gh[g], hz = gh[g + HDIM], hn = gh[g + 2 * HDIM];
  float r = 1.0f / (1.0f + __expf(-(ir + hr)));
  float z = 1.0f / (1.0f + __expf(-(iz + hz)));
  float n = tanhf(in + r * hn);
  float ho = h[(size_t)b * HDIM + j];
  float nh = (1.0f - z) * n + z * ho;
  if (plen && !(t < plen[b])) nh = ho;
  hout[(size_t)b * HDIM + j] = nh;
  hbf[(size_t)b * HDIM + j]  = (bf16_t)nh;
}

// pred_value[b,t,:] = (t < plen[b]) ? h1[b,:] : 0
__global__ void enc_store_out(const float* __restrict__ h1, float* __restrict__ pv,
                              const int* __restrict__ plen, int t) {
  int i = blockIdx.x * 256 + threadIdx.x;
  int b = i / HDIM, j = i % HDIM;
  if (b >= BSZ) return;
  pv[((size_t)b * T1 + t) * HDIM + j] = (t < plen[b]) ? h1[(size_t)b * HDIM + j] : 0.0f;
}

// per-row argmax over N with first-index tie break (jnp.argmax semantics)
__global__ void row_argmax(const float* __restrict__ logits, long ld, int N,
                           int* __restrict__ tokens, int* __restrict__ ptok, int t) {
  int b = blockIdx.x;
  __shared__ float sv[256]; __shared__ int si[256];
  float bv = -3.4e38f; int bi = N;
  for (int i = threadIdx.x; i < N; i += 256) {
    float v = logits[(size_t)b * ld + i];
    if (v > bv || (v == bv && i < bi)) { bv = v; bi = i; }
  }
  sv[threadIdx.x] = bv; si[threadIdx.x] = bi; __syncthreads();
  for (int o = 128; o > 0; o >>= 1) {
    if (threadIdx.x < o) {
      float ov = sv[threadIdx.x + o]; int oi = si[threadIdx.x + o];
      if (ov > sv[threadIdx.x] || (ov == sv[threadIdx.x] && oi < si[threadIdx.x])) {
        sv[threadIdx.x] = ov; si[threadIdx.x] = oi;
      }
    }
    __syncthreads();
  }
  if (threadIdx.x == 0) {
    tokens[b] = si[0];
    if (ptok) ptok[b * T1 + t] = si[0];
  }
}

// pred_len[b] = (last position of END in pred_token[b,:]) + 1
__global__ void compute_pred_len(const int* __restrict__ ptok, int* __restrict__ plen) {
  int b = threadIdx.x;
  if (b < BSZ) {
    int p = 0;
    for (int j = 0; j < T1; ++j) if (ptok[b * T1 + j] == END_TOK) p = j;
    plen[b] = p + 1;
  }
}

// ============================================================
// Host orchestration
// ============================================================
extern "C" void kernel_launch(void* const* d_in, const int* in_sizes, int n_in,
                              void* d_out, int out_size, void* d_ws, size_t ws_size,
                              hipStream_t stream) {
  (void)in_sizes; (void)n_in; (void)out_size; (void)ws_size;

  // ---- inputs (setup_inputs dict order, params flattened in insertion order)
  const float* in_hidden   = (const float*)d_in[0];   // [2,32,1024]
  const float* attn_value  = (const float*)d_in[1];   // [32,50,1024]
  const int*   attn_mask   = (const int*)  d_in[2];   // [32,50]
  /* d_in[3] = num_steps (device scalar; compile-time 50 used) */
  const float* W_emb  = (const float*)d_in[4];
  const float* d1Wih0 = (const float*)d_in[5];  const float* d1Whh0 = (const float*)d_in[6];
  const float* d1bih0 = (const float*)d_in[7];  const float* d1bhh0 = (const float*)d_in[8];
  const float* d1Wih1 = (const float*)d_in[9];  const float* d1Whh1 = (const float*)d_in[10];
  const float* d1bih1 = (const float*)d_in[11]; const float* d1bhh1 = (const float*)d_in[12];
  const float* out1W  = (const float*)d_in[13]; const float* out1b  = (const float*)d_in[14];
  const float* eWih0  = (const float*)d_in[15]; const float* eWhh0  = (const float*)d_in[16];
  const float* ebih0  = (const float*)d_in[17]; const float* ebhh0  = (const float*)d_in[18];
  const float* eWih1  = (const float*)d_in[19]; const float* eWhh1  = (const float*)d_in[20];
  const float* ebih1  = (const float*)d_in[21]; const float* ebhh1  = (const float*)d_in[22];
  const float* d2Wih0 = (const float*)d_in[23]; const float* d2Whh0 = (const float*)d_in[24];
  const float* d2bih0 = (const float*)d_in[25]; const float* d2bhh0 = (const float*)d_in[26];
  const float* d2Wih1 = (const float*)d_in[27]; const float* d2Whh1 = (const float*)d_in[28];
  const float* d2bih1 = (const float*)d_in[29]; const float* d2bhh1 = (const float*)d_in[30];
  const float* out2W  = (const float*)d_in[31]; const float* out2b  = (const float*)d_in[32];

  float* logits1 = (float*)d_out;                              // [32,50,16000] (B-major)
  float* logits2 = logits1 + (size_t)BSZ * TSTEPS * VSZ;       // [32,50,16000]
  const long LLD = (long)TSTEPS * VSZ;                         // batch-row stride of logits

  // ---- workspace carve-out (deterministic; 256B aligned)
  char* wp = (char*)d_ws;
  auto carve = [&](size_t bytes) { void* p = wp; wp += (bytes + 255) & ~(size_t)255; return p; };
  auto bfw = [&](size_t n) { return (bf16_t*)carve(n * sizeof(bf16_t)); };

  bf16_t* bEmb    = bfw((size_t)VSZ * EDIM);
  bf16_t* bd1Wih0 = bfw((size_t)G3 * (EDIM + HDIM));
  bf16_t* bd1Whh0 = bfw((size_t)G3 * HDIM);
  bf16_t* bd1Wih1 = bfw((size_t)G3 * HDIM);
  bf16_t* bd1Whh1 = bfw((size_t)G3 * HDIM);
  bf16_t* bout1W  = bfw((size_t)VSZ * HDIM);
  bf16_t* beWih0  = bfw((size_t)G3 * EDIM);
  bf16_t* beWhh0  = bfw((size_t)G3 * HDIM);
  bf16_t* beWih1  = bfw((size_t)G3 * HDIM);
  bf16_t* beWhh1  = bfw((size_t)G3 * HDIM);
  bf16_t* bd2Wih0 = bfw((size_t)G3 * (EDIM + 2 * HDIM));
  bf16_t* bd2Whh0 = bfw((size_t)G3 * HDIM);
  bf16_t* bd2Wih1 = bfw((size_t)G3 * HDIM);
  bf16_t* bd2Whh1 = bfw((size_t)G3 * HDIM);
  bf16_t* bout2W  = bfw((size_t)VSZ * HDIM);

  bf16_t* xbuf  = bfw((size_t)BSZ * (EDIM + 2 * HDIM)); // widest x (2560)
  bf16_t* h0bf  = bfw((size_t)BSZ * HDIM);
  bf16_t* h1bf  = bfw((size_t)BSZ * HDIM);
  float*  h0    = (float*)carve((size_t)BSZ * HDIM * 4);
  float*  h1    = (float*)carve((size_t)BSZ * HDIM * 4);
  float*  gi    = (float*)carve((size_t)BSZ * G3 * 4);
  float*  gh    = (float*)carve((size_t)BSZ * G3 * 4);
  float*  predv = (float*)carve((size_t)BSZ * T1 * HDIM * 4);
  int*    tokens = (int*)carve(BSZ * 4);
  int*    ptok   = (int*)carve(BSZ * T1 * 4);
  int*    plen   = (int*)carve(BSZ * 4);

  auto cvt = [&](const float* s, bf16_t* d, size_t n) {
    cvt_f32_bf16<<<dim3((unsigned)((n + 255) / 256)), 256, 0, stream>>>(s, d, (int)n);
  };
  auto gemm = [&](const bf16_t* A, int lda, const bf16_t* W, int ldw,
                  const float* bias, float* C, long ldc, int N, int K) {
    GemmArgs g{A, lda, W, ldw, bias, C, ldc, K};
    wmma_gemm_bias<<<dim3(N / 64, BSZ / 16), 128, 0, stream>>>(g);
  };
  // two independent N=3072 gate GEMMs in one launch
  auto gemm2 = [&](const bf16_t* A0, int lda0, const bf16_t* W0, int ldw0,
                   const float* b0, float* C0, int K0,
                   const bf16_t* A1, int lda1, const bf16_t* W1, int ldw1,
                   const float* b1, float* C1, int K1) {
    GemmArgs g0{A0, lda0, W0, ldw0, b0, C0, (long)G3, K0};
    GemmArgs g1{A1, lda1, W1, ldw1, b1, C1, (long)G3, K1};
    wmma_gemm_bias2<<<dim3(G3 / 64, BSZ / 16, 2), 128, 0, stream>>>(g0, g1);
  };
  const dim3 gBH(BSZ * HDIM / 256);
  auto prime = [&](const float* h, bf16_t* hb) {
    h_to_bf16<<<gBH, 256, 0, stream>>>(h, hb);
  };

  // ---- one-time (per launch) weight conversion to bf16 (L2-resident afterwards)
  cvt(W_emb,  bEmb,    (size_t)VSZ * EDIM);
  cvt(d1Wih0, bd1Wih0, (size_t)G3 * (EDIM + HDIM));
  cvt(d1Whh0, bd1Whh0, (size_t)G3 * HDIM);
  cvt(d1Wih1, bd1Wih1, (size_t)G3 * HDIM);
  cvt(d1Whh1, bd1Whh1, (size_t)G3 * HDIM);
  cvt(out1W,  bout1W,  (size_t)VSZ * HDIM);
  cvt(eWih0,  beWih0,  (size_t)G3 * EDIM);
  cvt(eWhh0,  beWhh0,  (size_t)G3 * HDIM);
  cvt(eWih1,  beWih1,  (size_t)G3 * HDIM);
  cvt(eWhh1,  beWhh1,  (size_t)G3 * HDIM);
  cvt(d2Wih0, bd2Wih0, (size_t)G3 * (EDIM + 2 * HDIM));
  cvt(d2Whh0, bd2Whh0, (size_t)G3 * HDIM);
  cvt(d2Wih1, bd2Wih1, (size_t)G3 * HDIM);
  cvt(d2Whh1, bd2Whh1, (size_t)G3 * HDIM);
  cvt(out2W,  bout2W,  (size_t)VSZ * HDIM);

  // ================= Phase 1: decoder 1 =================
  copy_f32<<<gBH, 256, 0, stream>>>(in_hidden,              h0, BSZ * HDIM);
  copy_f32<<<gBH, 256, 0, stream>>>(in_hidden + BSZ * HDIM, h1, BSZ * HDIM);
  prime(h0, h0bf); prime(h1, h1bf);
  set_int<<<1, 32, 0, stream>>>(tokens, START_TOK, BSZ);

  for (int t = 0; t < TSTEPS; ++t) {
    const int LDX = EDIM + HDIM;  // 1536
    embed_bf16<<<dim3(BSZ * EDIM / 256), 256, 0, stream>>>(bEmb, tokens, 1, xbuf, LDX);
    fused_attend<<<BSZ, 256, 0, stream>>>(h1, attn_value, attn_mask, nullptr, RLEN,
                                          xbuf, LDX, EDIM);
    // layer 0: gi = x@Wih^T, gh = h0@Whh^T in one launch
    gemm2(xbuf, LDX,  bd1Wih0, LDX,  d1bih0, gi, LDX,
          h0bf, HDIM, bd1Whh0, HDIM, d1bhh0, gh, HDIM);
    gru_gates<<<gBH, 256, 0, stream>>>(gi, gh, h0, h0, h0bf, nullptr, 0);
    // layer 1
    gemm2(h0bf, HDIM, bd1Wih1, HDIM, d1bih1, gi, HDIM,
          h1bf, HDIM, bd1Whh1, HDIM, d1bhh1, gh, HDIM);
    gru_gates<<<gBH, 256, 0, stream>>>(gi, gh, h1, h1, h1bf, nullptr, 0);
    // vocab projection straight into d_out [B, T, V]
    gemm(h1bf, HDIM, bout1W, HDIM, out1b, logits1 + (size_t)t * VSZ, LLD, VSZ, HDIM);
    row_argmax<<<BSZ, 256, 0, stream>>>(logits1 + (size_t)t * VSZ, LLD, VSZ, tokens, ptok, t);
  }

  set_last_end<<<1, 32, 0, stream>>>(ptok);
  compute_pred_len<<<1, 32, 0, stream>>>(ptok, plen);

  // ================= Phase 2: encoder over predicted tokens =================
  zero_f32<<<gBH, 256, 0, stream>>>(h0, BSZ * HDIM);
  zero_f32<<<gBH, 256, 0, stream>>>(h1, BSZ * HDIM);
  prime(h0, h0bf); prime(h1, h1bf);

  for (int t = 0; t < T1; ++t) {
    embed_bf16<<<dim3(BSZ * EDIM / 256), 256, 0, stream>>>(bEmb, ptok + t, T1, xbuf, EDIM);
    gemm2(xbuf, EDIM, beWih0, EDIM, ebih0, gi, EDIM,
          h0bf, HDIM, beWhh0, HDIM, ebhh0, gh, HDIM);
    gru_gates<<<gBH, 256, 0, stream>>>(gi, gh, h0, h0, h0bf, plen, t);
    gemm2(h0bf, HDIM, beWih1, HDIM, ebih1, gi, HDIM,
          h1bf, HDIM, beWhh1, HDIM, ebhh1, gh, HDIM);
    gru_gates<<<gBH, 256, 0, stream>>>(gi, gh, h1, h1, h1bf, plen, t);
    enc_store_out<<<gBH, 256, 0, stream>>>(h1, predv, plen, t);
  }

  // ================= Phase 3: decoder 2 =================
  set_int<<<1, 32, 0, stream>>>(tokens, START_TOK, BSZ);
  // h0/h1 (+bf16 copies) carry over from encoder final state

  for (int t = 0; t < TSTEPS; ++t) {
    const int LDX = EDIM + 2 * HDIM;  // 2560
    embed_bf16<<<dim3(BSZ * EDIM / 256), 256, 0, stream>>>(bEmb, tokens, 1, xbuf, LDX);
    fused_attend<<<BSZ, 256, 0, stream>>>(h1, attn_value, attn_mask, nullptr, RLEN,
                                          xbuf, LDX, EDIM);
    fused_attend<<<BSZ, 256, 0, stream>>>(h1, predv, nullptr, plen, T1,
                                          xbuf, LDX, EDIM + HDIM);
    gemm2(xbuf, LDX,  bd2Wih0, LDX,  d2bih0, gi, LDX,
          h0bf, HDIM, bd2Whh0, HDIM, d2bhh0, gh, HDIM);
    gru_gates<<<gBH, 256, 0, stream>>>(gi, gh, h0, h0, h0bf, nullptr, 0);
    gemm2(h0bf, HDIM, bd2Wih1, HDIM, d2bih1, gi, HDIM,
          h1bf, HDIM, bd2Whh1, HDIM, d2bhh1, gh, HDIM);
    gru_gates<<<gBH, 256, 0, stream>>>(gi, gh, h1, h1, h1bf, nullptr, 0);
    gemm(h1bf, HDIM, bout2W, HDIM, out2b, logits2 + (size_t)t * VSZ, LLD, VSZ, HDIM);
    row_argmax<<<BSZ, 256, 0, stream>>>(logits2 + (size_t)t * VSZ, LLD, VSZ, tokens, nullptr, t);
  }
}
